// Attention_39573828665647
// MI455X (gfx1250) — matrix operand order repaired and checked by reading, not compiled, not run on hardware
//
#include <hip/hip_runtime.h>

// Problem constants (match reference)
#define B_    4
#define S_    1024
#define DIM_  2048
#define NH_   32
#define NKV_  8
#define HD_   64
#define ATTN_ 3072      // (32 + 2*8) * 64
#define MROWS_ 4096     // B*S

typedef __attribute__((ext_vector_type(16))) __bf16 v16bf;
typedef __attribute__((ext_vector_type(8)))  float  v8f;
typedef __attribute__((ext_vector_type(4)))  unsigned int u32x4;

union Frag {
  v16bf v;
  u32x4 q[2];
};

static __device__ __forceinline__ unsigned short f2bf(float f) {
  unsigned int u = __builtin_bit_cast(unsigned int, f);
  u += 0x7fffu + ((u >> 16) & 1u);          // round-to-nearest-even
  return (unsigned short)(u >> 16);
}
static __device__ __forceinline__ float bf2f(unsigned short h) {
  unsigned int u = ((unsigned int)h) << 16;
  return __builtin_bit_cast(float, u);
}
static __device__ __forceinline__ v8f vzero8() {
  v8f z;
#pragma unroll
  for (int i = 0; i < 8; ++i) z[i] = 0.0f;
  return z;
}
static __device__ __forceinline__ v8f wmma_bf16(v16bf a, v16bf b, v8f c) {
  // (neg_a, A, neg_b, B, c_mod, C, reuse_a, reuse_b)
  return __builtin_amdgcn_wmma_f32_16x16x32_bf16(false, a, false, b, (short)0, c,
                                                 false, false);
}

// LDS byte offset of a __shared__ object: flat LDS aperture maps addr[31:0]
// directly to the wave's LDS offset (ISA 10.2 per-aperture calc).
static __device__ __forceinline__ unsigned lds_off32(const void* p) {
  return (unsigned)(unsigned long long)p;
}
// Async global -> LDS 16-byte copy (per lane), tracked with ASYNCcnt.
static __device__ __forceinline__ void async_ld_b128(unsigned lds_off,
                                                     const unsigned short* gptr) {
  asm volatile("global_load_async_to_lds_b128 %0, %1, off"
               :: "v"(lds_off), "v"(gptr)
               : "memory");
}

// ---------------------------------------------------------------- convert
__global__ __launch_bounds__(256) void f32_to_bf16_kernel(
    const float* __restrict__ src, unsigned short* __restrict__ dst, int n) {
  int i = blockIdx.x * blockDim.x + threadIdx.x;
  if (i < n) dst[i] = f2bf(src[i]);
}

// ---------------------------------------------------------------- GEMM
// C[M,N] = A[M,K] * B[N,K]^T ; A,B bf16 row-major, C f32 or bf16.
// Block: 128(M) x 128(N), 8 waves (2M x 4N), each wave 64x32 = 4x2 WMMA tiles.
// K-step 32, double-buffered async global->LDS staging of A and B tiles.
#define KPAD 40   // 32 + 8 pad ushorts -> 80B row pitch, conflict-free b128 reads

__global__ __launch_bounds__(256) void gemm_bf16_kernel(
    const unsigned short* __restrict__ A, const unsigned short* __restrict__ Bm,
    void* __restrict__ C, int M, int N, int K, int out_bf16) {
  __shared__ __attribute__((aligned(16))) unsigned short As[2][128][KPAD];
  __shared__ __attribute__((aligned(16))) unsigned short Bs[2][128][KPAD];

  const int tid  = threadIdx.x;
  const int lane = tid & 31;
  const int wave = tid >> 5;
  const int hlf  = lane >> 4;
  const int l16  = lane & 15;
  const int m_blk = blockIdx.y * 128;
  const int n_blk = blockIdx.x * 128;
  const int wm = (wave & 1) * 64;   // wave's M offset inside the block tile
  const int wn = (wave >> 1) * 32;  // wave's N offset inside the block tile

  // Issue one 128x32 A tile + one 128x32 B tile into LDS buffer p (async).
  // 512 16B chunks per tile, 2 per thread per matrix -> 4 async instr / wave.
  auto issue_tile = [&](int p, int k0) {
#pragma unroll
    for (int c = 0; c < 2; ++c) {
      const int chunk = tid + c * 256;
      const int row = chunk >> 2;
      const int col = (chunk & 3) * 8;   // ushort units (16B chunks)
      async_ld_b128(lds_off32(&As[p][row][col]),
                    A + (size_t)(m_blk + row) * K + k0 + col);
      async_ld_b128(lds_off32(&Bs[p][row][col]),
                    Bm + (size_t)(n_blk + row) * K + k0 + col);
    }
  };

  v8f acc[4][2];
#pragma unroll
  for (int i = 0; i < 4; ++i)
#pragma unroll
    for (int j = 0; j < 2; ++j) acc[i][j] = vzero8();

  // prologue: two tiles in flight
  issue_tile(0, 0);
  issue_tile(1, 32);

  int p = 0;
  for (int k0 = 0; k0 < K; k0 += 32, p ^= 1) {
    if (k0 + 32 < K)
      asm volatile("s_wait_asynccnt 0x4" ::: "memory");   // oldest tile landed
    else
      asm volatile("s_wait_asynccnt 0x0" ::: "memory");   // drain
    __syncthreads();                                      // all waves' data in LDS

    Frag a[4], b[2];
#pragma unroll
    for (int i = 0; i < 4; ++i) {
      // A frag: lane = row M; elems 0..7 -> k = hlf*8 + e ; 8..15 -> +16
      const unsigned short* q = &As[p][wm + i * 16 + l16][hlf * 8];
      a[i].q[0] = *(const u32x4*)q;
      a[i].q[1] = *(const u32x4*)(q + 16);
    }
#pragma unroll
    for (int j = 0; j < 2; ++j) {
      // B frag: lane = col N; elems 0..7 -> k = hlf*8 + e ; 8..15 -> +16
      const unsigned short* q = &Bs[p][wn + j * 16 + l16][hlf * 8];
      b[j].q[0] = *(const u32x4*)q;
      b[j].q[1] = *(const u32x4*)(q + 16);
    }
#pragma unroll
    for (int i = 0; i < 4; ++i)
#pragma unroll
      for (int j = 0; j < 2; ++j)
        acc[i][j] = wmma_bf16(a[i].v, b[j].v, acc[i][j]);

    __syncthreads();                         // everyone done reading buffer p
    if (k0 + 64 < K) issue_tile(p, k0 + 64); // refill p for iteration k0+64
  }

#pragma unroll
  for (int i = 0; i < 4; ++i)
#pragma unroll
    for (int j = 0; j < 2; ++j)
#pragma unroll
      for (int r = 0; r < 8; ++r) {
        const int row = m_blk + wm + i * 16 + hlf * 8 + r;  // C layout: VGPR r
        const int col = n_blk + wn + j * 16 + l16;
        const size_t idx = (size_t)row * N + col;
        if (out_bf16) ((unsigned short*)C)[idx] = f2bf(acc[i][j][r]);
        else          ((float*)C)[idx]          = acc[i][j][r];
      }
}

// ---------------------------------------------------------------- RoPE split
// qkv[M, 3072] bf16 -> Q[b,h,s,hd] (scaled by 1/8), K[b,kv,s,hd], V^T[b,kv,hd,s]
__global__ __launch_bounds__(256) void rope_split_kernel(
    const unsigned short* __restrict__ qkv, const float* __restrict__ fcos,
    const float* __restrict__ fsin, unsigned short* __restrict__ Q,
    unsigned short* __restrict__ K, unsigned short* __restrict__ V, int n) {
  int idx = blockIdx.x * blockDim.x + threadIdx.x;
  if (idx >= n) return;
  const int PAIRS = ATTN_ / 2;               // 1536 pairs per row
  const int c   = idx % PAIRS;
  const int row = idx / PAIRS;               // b*S + s
  const int s = row & (S_ - 1);
  const int b = row >> 10;
  const int a0  = c * 2;
  const int kv  = a0 / (6 * HD_);            // 6 slots of 64 per kv group
  const int rem = a0 - kv * (6 * HD_);
  const int j   = rem >> 6;                  // 0..3 q heads, 4 = k, 5 = v
  const int hd  = rem & 63;                  // even
  const int i   = hd >> 1;

  const float xr = bf2f(qkv[(size_t)row * ATTN_ + a0]);
  const float xi = bf2f(qkv[(size_t)row * ATTN_ + a0 + 1]);

  if (j < 5) {
    const float cs = fcos[s * (HD_ / 2) + i];
    const float sn = fsin[s * (HD_ / 2) + i];
    const float orr = xr * cs - xi * sn;
    const float oii = xr * sn + xi * cs;
    if (j < 4) {
      const int head = kv * 4 + j;
      const float qs = 0.125f;               // fold 1/sqrt(HD) into Q
      unsigned short* d = Q + ((size_t)(b * NH_ + head) * S_ + s) * HD_ + hd;
      d[0] = f2bf(orr * qs);
      d[1] = f2bf(oii * qs);
    } else {
      unsigned short* d = K + ((size_t)(b * NKV_ + kv) * S_ + s) * HD_ + hd;
      d[0] = f2bf(orr);
      d[1] = f2bf(oii);
    }
  } else {
    unsigned short* d = V + ((size_t)(b * NKV_ + kv) * HD_ + hd) * S_ + s;
    d[0]   = f2bf(xr);                       // row hd
    d[S_]  = f2bf(xi);                       // row hd+1
  }
}

// ---------------------------------------------------------------- flash attn
// Grid: B*NH*(S/128) blocks, 256 threads. Wave w owns 16 query rows.
// NOTE: waves have different causal trip counts -> no block-wide barriers here;
// keep direct global b128 loads + wave-private LDS transpose staging.
__global__ __launch_bounds__(256) void flash_attn_kernel(
    const unsigned short* __restrict__ Q, const unsigned short* __restrict__ K,
    const unsigned short* __restrict__ V, unsigned short* __restrict__ O) {
  __shared__ __attribute__((aligned(16))) unsigned short plds[8][16][32];
  const int lane = threadIdx.x & 31;
  const int wave = threadIdx.x >> 5;
  const int hlf  = lane >> 4;
  const int l16  = lane & 15;
  const int bid  = blockIdx.x;
  const int qblk = bid & 7;                  // S/128 = 8
  const int h    = (bid >> 3) & 31;
  const int b    = bid >> 8;
  const int kvh  = h >> 2;                   // GQA 4:1
  const unsigned short* qbase = Q + (size_t)(b * NH_ + h) * S_ * HD_;
  const unsigned short* kbase = K + (size_t)(b * NKV_ + kvh) * S_ * HD_;
  const unsigned short* vbase = V + (size_t)(b * NKV_ + kvh) * HD_ * S_;  // [hd][s]
  const int m0 = qblk * 128 + wave * 16;

  Frag qa[2];                                // Q as A-frags, K-dim = HD (2 chunks)
#pragma unroll
  for (int c = 0; c < 2; ++c) {
    const unsigned short* p = qbase + (size_t)(m0 + l16) * HD_ + c * 32 + hlf * 8;
    qa[c].q[0] = *(const u32x4*)p;
    qa[c].q[1] = *(const u32x4*)(p + 16);
  }

  v8f o[4];
#pragma unroll
  for (int t = 0; t < 4; ++t) o[t] = vzero8();
  float mi[8], li[8];
#pragma unroll
  for (int r = 0; r < 8; ++r) { mi[r] = -1e30f; li[r] = 0.0f; }

  for (int j0 = 0; j0 < m0 + 16; j0 += 32) {   // causal bound
    // --- scores: two 16x16 tiles (keys j0..+15, j0+16..+31)
    Frag kb[2][2];
#pragma unroll
    for (int t = 0; t < 2; ++t)
#pragma unroll
      for (int c = 0; c < 2; ++c) {
        const unsigned short* p =
            kbase + (size_t)(j0 + t * 16 + l16) * HD_ + c * 32 + hlf * 8;
        kb[t][c].q[0] = *(const u32x4*)p;
        kb[t][c].q[1] = *(const u32x4*)(p + 16);
      }
    v8f s0 = wmma_bf16(qa[0].v, kb[0][0].v, vzero8());
    s0 = wmma_bf16(qa[1].v, kb[0][1].v, s0);
    v8f s1 = wmma_bf16(qa[0].v, kb[1][0].v, vzero8());
    s1 = wmma_bf16(qa[1].v, kb[1][1].v, s1);

    // --- online softmax (rows live per half-wave; reduce across 16 lanes)
    const int key0 = j0 + l16;
    const int key1 = key0 + 16;
    float p0v[8], p1v[8], alpha[8];
#pragma unroll
    for (int r = 0; r < 8; ++r) {
      const int qrow = m0 + hlf * 8 + r;
      float v0 = (key0 <= qrow) ? s0[r] : -1e30f;
      float v1 = (key1 <= qrow) ? s1[r] : -1e30f;
      float mx = fmaxf(v0, v1);
#pragma unroll
      for (int off = 8; off; off >>= 1) mx = fmaxf(mx, __shfl_xor(mx, off, 16));
      const float nm = fmaxf(mi[r], mx);
      const float al = __expf(mi[r] - nm);
      const float e0 = __expf(v0 - nm);
      const float e1 = __expf(v1 - nm);
      float rs = e0 + e1;
#pragma unroll
      for (int off = 8; off; off >>= 1) rs += __shfl_xor(rs, off, 16);
      li[r] = li[r] * al + rs;
      mi[r] = nm;
      alpha[r] = al;
      p0v[r] = e0;
      p1v[r] = e1;
    }
#pragma unroll
    for (int t = 0; t < 4; ++t)
#pragma unroll
      for (int r = 0; r < 8; ++r) o[t][r] *= alpha[r];

    // --- redistribute P (C-layout, lane=col) -> A-layout (lane=row) via LDS
#pragma unroll
    for (int r = 0; r < 8; ++r) {
      const int prow = hlf * 8 + r;
      plds[wave][prow][l16]      = f2bf(p0v[r]);
      plds[wave][prow][l16 + 16] = f2bf(p1v[r]);
    }
    asm volatile("s_wait_dscnt 0x0" ::: "memory");  // wave-private, DS in-order
    Frag pa;
    {
      const unsigned short* p = &plds[wave][l16][hlf * 8];
      pa.q[0] = *(const u32x4*)p;
      pa.q[1] = *(const u32x4*)(p + 16);
    }

    // --- O += P (16x32) * V (32x64): V^T rows are contiguous in key
#pragma unroll
    for (int t = 0; t < 4; ++t) {
      Frag vb;
      const unsigned short* p = vbase + (size_t)(t * 16 + l16) * S_ + j0 + hlf * 8;
      vb.q[0] = *(const u32x4*)p;
      vb.q[1] = *(const u32x4*)(p + 16);
      o[t] = wmma_bf16(pa.v, vb.v, o[t]);
    }
  }

  // epilogue: normalize and store bf16 [B,S,DIM]
#pragma unroll
  for (int t = 0; t < 4; ++t)
#pragma unroll
    for (int r = 0; r < 8; ++r) {
      const int row = m0 + hlf * 8 + r;
      const float val = o[t][r] / li[r];
      O[((size_t)b * S_ + row) * DIM_ + h * HD_ + t * 16 + l16] = f2bf(val);
    }
}

// ---------------------------------------------------------------- launch
extern "C" void kernel_launch(void* const* d_in, const int* in_sizes, int n_in,
                              void* d_out, int out_size, void* d_ws, size_t ws_size,
                              hipStream_t stream) {
  (void)in_sizes; (void)n_in; (void)out_size; (void)ws_size;
  const float* x    = (const float*)d_in[0];
  const float* fcos = (const float*)d_in[1];
  const float* fsin = (const float*)d_in[2];
  const float* wqkv = (const float*)d_in[3];
  const float* wo   = (const float*)d_in[4];

  char* ws = (char*)d_ws;
  size_t off = 0;
  auto wsAlloc = [&](size_t bytes) -> unsigned short* {
    unsigned short* p = (unsigned short*)(ws + off);
    off += (bytes + 255) & ~(size_t)255;
    return p;
  };
  unsigned short* xb   = wsAlloc((size_t)MROWS_ * DIM_ * 2);          // 16 MB
  unsigned short* wqb  = wsAlloc((size_t)ATTN_ * DIM_ * 2);           // 12 MB
  unsigned short* wob  = wsAlloc((size_t)DIM_ * DIM_ * 2);            //  8 MB
  unsigned short* qkvb = wsAlloc((size_t)MROWS_ * ATTN_ * 2);         // 24 MB
  unsigned short* qws  = wsAlloc((size_t)B_ * NH_ * S_ * HD_ * 2);    // 16 MB
  unsigned short* kws  = wsAlloc((size_t)B_ * NKV_ * S_ * HD_ * 2);   //  4 MB
  unsigned short* vws  = wsAlloc((size_t)B_ * NKV_ * HD_ * S_ * 2);   //  4 MB
  unsigned short* ows  = wsAlloc((size_t)MROWS_ * DIM_ * 2);          // 16 MB

  const int nx  = MROWS_ * DIM_;   // 8388608
  const int nwq = ATTN_ * DIM_;    // 6291456
  const int nwo = DIM_ * DIM_;     // 4194304
  f32_to_bf16_kernel<<<(nx  + 255) / 256, 256, 0, stream>>>(x, xb, nx);
  f32_to_bf16_kernel<<<(nwq + 255) / 256, 256, 0, stream>>>(wqkv, wqb, nwq);
  f32_to_bf16_kernel<<<(nwo + 255) / 256, 256, 0, stream>>>(wo, wob, nwo);

  // QKV GEMM: [4096,2048] x [3072,2048]^T -> bf16 [4096,3072]
  gemm_bf16_kernel<<<dim3(ATTN_ / 128, MROWS_ / 128), 256, 0, stream>>>(
      xb, wqb, qkvb, MROWS_, ATTN_, DIM_, 1);

  const int nrope = B_ * S_ * (ATTN_ / 2);   // 6291456
  rope_split_kernel<<<(nrope + 255) / 256, 256, 0, stream>>>(
      qkvb, fcos, fsin, qws, kws, vws, nrope);

  flash_attn_kernel<<<B_ * NH_ * (S_ / 128), 256, 0, stream>>>(qws, kws, vws, ows);

  // Output GEMM: [4096,2048] x [2048,2048]^T -> f32 d_out
  gemm_bf16_kernel<<<dim3(DIM_ / 128, MROWS_ / 128), 256, 0, stream>>>(
      ows, wob, d_out, MROWS_, DIM_, DIM_, 0);
}
